// SimpleGraphDenoiser_38663295598682
// MI455X (gfx1250) — matrix operand order repaired and checked by reading, not compiled, not run on hardware
//
#include <hip/hip_runtime.h>

#define B_    2
#define V_    512
#define HID_  64
#define EH_   128
#define EPS_  1e-5f
#define TDIM_ 32

typedef float        v8f  __attribute__((ext_vector_type(8)));
typedef float        v2f  __attribute__((ext_vector_type(2)));
typedef _Float16     v16h __attribute__((ext_vector_type(16)));
typedef unsigned int v4u  __attribute__((ext_vector_type(4)));
typedef int          v8i  __attribute__((ext_vector_type(8)));
typedef int          v4i  __attribute__((ext_vector_type(4)));

__device__ __forceinline__ v8f wmma_f32_k4(v2f a, v2f b, v8f c) {
  // V_WMMA_F32_16X16X4_F32 : exact f32 16x16 tile, K=4
  return __builtin_amdgcn_wmma_f32_16x16x4_f32(false, a, false, b, (short)0, c, false, false);
}
__device__ __forceinline__ v8f wmma_f16_k32(v16h a, v16h b, v8f c) {
  // V_WMMA_F32_16X16X32_F16 : f16 inputs, f32 accumulate
  return __builtin_amdgcn_wmma_f32_16x16x32_f16(false, a, false, b, (short)0, c, false, false);
}

// ---------------------------------------------------------------------------
// K1: time embedding MLP -> t_emb (B x 64);  T = t_emb @ eW1[133:197] + eb1
//     (B x 128); f16 copy of eW2.  One block of 256 threads.
// ---------------------------------------------------------------------------
__global__ void k_setup(const float* __restrict__ t,
                        const float* __restrict__ tW1, const float* __restrict__ tb1,
                        const float* __restrict__ tW2, const float* __restrict__ tb2,
                        const float* __restrict__ eW1, const float* __restrict__ eb1,
                        const float* __restrict__ eW2,
                        float* __restrict__ temb, float* __restrict__ tvec,
                        _Float16* __restrict__ w2h) {
  __shared__ float sf[B_][TDIM_];
  __shared__ float sx[B_][HID_];
  __shared__ float st[B_][HID_];
  int id = threadIdx.x;
  if (id < B_ * TDIM_) {
    int b = id / TDIM_, i = id % TDIM_;
    int i2 = i & 15;
    float fr  = expf(-logf(10000.f) * (float)i2 / 16.f);
    float arg = t[b] * fr;
    sf[b][i] = (i < 16) ? sinf(arg) : cosf(arg);
  }
  __syncthreads();
  if (id < B_ * HID_) {
    int b = id / HID_, j = id % HID_;
    float acc = tb1[j];
    for (int k = 0; k < TDIM_; ++k) acc += sf[b][k] * tW1[k * HID_ + j];
    sx[b][j] = acc / (1.f + expf(-acc));   // silu
  }
  __syncthreads();
  if (id < B_ * HID_) {
    int b = id / HID_, j = id % HID_;
    float acc = tb2[j];
    for (int k = 0; k < HID_; ++k) acc += sx[b][k] * tW2[k * HID_ + j];
    temb[b * HID_ + j] = acc;
    st[b][j] = acc;
  }
  __syncthreads();
  if (id < B_ * EH_) {
    int b = id / EH_, j = id % EH_;
    float acc = eb1[j];
    for (int k = 0; k < HID_; ++k) acc += st[b][k] * eW1[(133 + k) * EH_ + j];
    tvec[b * EH_ + j] = acc;
  }
  for (int i = id; i < EH_ * 4; i += blockDim.x) w2h[i] = (_Float16)eW2[i];
}

// ---------------------------------------------------------------------------
// K2: h0[b,v,j] = nf[b,v]*npW[j] + npb[j] + t_emb[b,j]
// ---------------------------------------------------------------------------
__global__ void k_hinit(const float* __restrict__ nf, const float* __restrict__ npW,
                        const float* __restrict__ npb, const float* __restrict__ temb,
                        float* __restrict__ h0) {
  int idx = blockIdx.x * blockDim.x + threadIdx.x;   // B*V*HID
  int j = idx & (HID_ - 1);
  int v = (idx >> 6) & (V_ - 1);
  int b = idx >> 15;
  h0[idx] = nf[b * V_ + v] * npW[j] + npb[j] + temb[b * HID_ + j];
}

// ---------------------------------------------------------------------------
// K3: dis[b,u] = rsqrt(max(sum_v A[b,u,v], 1))
// ---------------------------------------------------------------------------
__global__ void k_dis(const float* __restrict__ A, float* __restrict__ dis) {
  __shared__ float red[128];
  int row = blockIdx.x;  // b*V+u
  const float* ar = A + (size_t)row * V_;
  float s = 0.f;
  for (int v = threadIdx.x; v < V_; v += 128) s += ar[v];
  red[threadIdx.x] = s;
  __syncthreads();
  for (int off = 64; off > 0; off >>= 1) {
    if (threadIdx.x < off) red[threadIdx.x] += red[threadIdx.x + off];
    __syncthreads();
  }
  if (threadIdx.x == 0) dis[row] = rsqrtf(fmaxf(red[0], 1.f));
}

// ---------------------------------------------------------------------------
// K4: An = dis[u] * A * dis[v]   (precomputed once, reused for 3 layers)
// ---------------------------------------------------------------------------
__global__ void k_anorm(const float* __restrict__ A, const float* __restrict__ dis,
                        float* __restrict__ An) {
  int idx = blockIdx.x * blockDim.x + threadIdx.x;  // B*V*V
  int v = idx & (V_ - 1);
  int u = (idx >> 9) & (V_ - 1);
  int b = idx >> 18;
  An[idx] = A[idx] * dis[b * V_ + u] * dis[b * V_ + v];
}

// ---------------------------------------------------------------------------
// K5: one GNN layer.  Block = 16-row u-strip of one batch, 128 thr (4 waves),
//     each wave owns one 16-wide N tile of HID=64.
//     neigh = An @ h (f32 WMMA, K=512), then out = h@Ws + neigh@Wn + biases
//     (f32 WMMA, K=64, neigh routed through LDS), relu, layernorm.
// ---------------------------------------------------------------------------
__global__ void k_gnn(const float* __restrict__ hin, float* __restrict__ hout,
                      const float* __restrict__ An,
                      const float* __restrict__ Ws, const float* __restrict__ bs,
                      const float* __restrict__ Wn, const float* __restrict__ bn,
                      const float* __restrict__ G,  const float* __restrict__ Bt) {
  __shared__ float sN[16][HID_ + 4];
  __shared__ float sO[16][HID_ + 4];
  __shared__ float sMu[16], sRs[16];
  int b   = blockIdx.x >> 5;
  int u0  = (blockIdx.x & 31) * 16;
  int tid = threadIdx.x;
  int wave = tid >> 5, lane = tid & 31;
  int hi = lane >> 4, col = lane & 15;
  int n0 = wave * 16;
  int koff = hi * 2;

  // phase 1: neigh tile = An[u-strip] @ hin   (M=16, N=16, K=512)
  v8f c = {};
  const float* Abase = An + ((size_t)(b * V_ + u0 + col)) * V_;
  for (int k = 0; k < V_; k += 4) {
    v2f a, bm;
    a.x = Abase[k + koff];
    a.y = Abase[k + koff + 1];
    const float* hp = hin + ((size_t)(b * V_ + k + koff)) * HID_ + n0 + col;
    bm.x = hp[0];
    bm.y = hp[HID_];
    c = wmma_f32_k4(a, bm, c);
  }
#pragma unroll
  for (int r = 0; r < 8; ++r) sN[r + hi * 8][n0 + col] = c[r];
  __syncthreads();

  // phase 2: out = h@Ws + neigh@Wn + (bs+bn)
  float bias = bs[n0 + col] + bn[n0 + col];
  v8f o = {bias, bias, bias, bias, bias, bias, bias, bias};
  const float* hrow = hin + ((size_t)(b * V_ + u0 + col)) * HID_;
  for (int k = 0; k < HID_; k += 4) {
    v2f a1, b1, a2, b2;
    a1.x = hrow[k + koff];
    a1.y = hrow[k + koff + 1];
    b1.x = Ws[(k + koff) * HID_ + n0 + col];
    b1.y = Ws[(k + koff + 1) * HID_ + n0 + col];
    o = wmma_f32_k4(a1, b1, o);
    a2.x = sN[col][k + koff];
    a2.y = sN[col][k + koff + 1];
    b2.x = Wn[(k + koff) * HID_ + n0 + col];
    b2.y = Wn[(k + koff + 1) * HID_ + n0 + col];
    o = wmma_f32_k4(a2, b2, o);
  }
#pragma unroll
  for (int r = 0; r < 8; ++r) sO[r + hi * 8][n0 + col] = fmaxf(o[r], 0.f);
  __syncthreads();

  // layernorm over HID per row
  if (tid < 16) {
    float mu = 0.f;
    for (int nn = 0; nn < HID_; ++nn) mu += sO[tid][nn];
    mu *= (1.f / HID_);
    float var = 0.f;
    for (int nn = 0; nn < HID_; ++nn) { float d = sO[tid][nn] - mu; var += d * d; }
    var *= (1.f / HID_);
    sMu[tid] = mu;
    sRs[tid] = rsqrtf(var + EPS_);
  }
  __syncthreads();
  for (int e = tid; e < 16 * HID_; e += blockDim.x) {
    int mm = e >> 6, nn = e & (HID_ - 1);
    hout[((size_t)(b * V_ + u0 + mm)) * HID_ + nn] =
        (sO[mm][nn] - sMu[mm]) * sRs[mm] * G[nn] + Bt[nn];
  }
}

// ---------------------------------------------------------------------------
// K6: P = h @ eW1[0:64]  (+ Tvec folded in),  Q = h @ eW1[64:128]
//     Block = 16-row strip, 256 thr (8 waves) cover N=128.
// ---------------------------------------------------------------------------
__global__ void k_pq(const float* __restrict__ h, const float* __restrict__ eW1,
                     const float* __restrict__ tvec,
                     float* __restrict__ P, float* __restrict__ Q) {
  int b  = blockIdx.x >> 5;
  int u0 = (blockIdx.x & 31) * 16;
  int tid = threadIdx.x;
  int wave = tid >> 5, lane = tid & 31;
  int hi = lane >> 4, col = lane & 15;
  int n0 = wave * 16;
  int koff = hi * 2;
  float tv = tvec[b * EH_ + n0 + col];
  v8f cP = {tv, tv, tv, tv, tv, tv, tv, tv};
  v8f cQ = {};
  const float* hrow = h + ((size_t)(b * V_ + u0 + col)) * HID_;
  for (int k = 0; k < HID_; k += 4) {
    v2f a, bP, bQ;
    a.x = hrow[k + koff];
    a.y = hrow[k + koff + 1];
    bP.x = eW1[(k + koff) * EH_ + n0 + col];
    bP.y = eW1[(k + koff + 1) * EH_ + n0 + col];
    bQ.x = eW1[(HID_ + k + koff) * EH_ + n0 + col];
    bQ.y = eW1[(HID_ + k + koff + 1) * EH_ + n0 + col];
    cP = wmma_f32_k4(a, bP, cP);
    cQ = wmma_f32_k4(a, bQ, cQ);
  }
#pragma unroll
  for (int r = 0; r < 8; ++r) {
    size_t oidx = ((size_t)(b * V_ + u0 + r + hi * 8)) * EH_ + n0 + col;
    P[oidx] = cP[r];
    Q[oidx] = cQ[r];
  }
}

// ---------------------------------------------------------------------------
// K7: edge MLP.  Block = (b, 16-v tile, 128-u chunk), 256 thr (8 waves);
//     each wave walks 16 u values.  Q tile staged to LDS by the Tensor Data
//     Mover (TDM) when available (pad_enable reproduces the 132-dword padded
//     rows), act(16x128) built per-lane in the f16 A-fragment layout with
//     b128 loads, logits = relu(act) @ eW2 via 4x WMMA f16 (eb2 in C).
// ---------------------------------------------------------------------------
#define ACT4(P4, Q4, R4, O4, D0)                                           \
  af[(D0) + 0] = (_Float16)fmaxf(P4.x + Q4.x + auv * R4.x + O4.x, 0.f);    \
  af[(D0) + 1] = (_Float16)fmaxf(P4.y + Q4.y + auv * R4.y + O4.y, 0.f);    \
  af[(D0) + 2] = (_Float16)fmaxf(P4.z + Q4.z + auv * R4.z + O4.z, 0.f);    \
  af[(D0) + 3] = (_Float16)fmaxf(P4.w + Q4.w + auv * R4.w + O4.w, 0.f);

__global__ void k_edge(const float* __restrict__ A, const int* __restrict__ obs,
                       const float* __restrict__ P, const float* __restrict__ Q,
                       const float* __restrict__ eW1,
                       const _Float16* __restrict__ w2h,
                       const float* __restrict__ eb2, float* __restrict__ out) {
  __shared__ __align__(16) float sQ[16][EH_ + 4];   // 132-float rows: 16B-aligned
  __shared__ __align__(16) float sRe[EH_];
  __shared__ __align__(16) float sObs[4][EH_];
  int bi = blockIdx.x;
  int b  = bi >> 7;
  int vt = (bi & 127) >> 2;
  int uc = bi & 3;
  int v0 = vt * 16;
  int tid = threadIdx.x;  // 256
  int wave = tid >> 5, lane = tid & 31;
  int hi = lane >> 4, col = lane & 15;

#if __has_builtin(__builtin_amdgcn_tensor_load_to_lds) && \
    __has_builtin(__builtin_amdgcn_s_wait_tensorcnt)
  // TDM: one wave issues a 2D tile descriptor; 16 rows x 128 dwords from Q,
  // padded +4 dwords every 128 (pad_interval=6 -> 128 dw, pad_amount=3 -> 4 dw)
  // to match the sQ[16][132] LDS layout.  Tracked by TENSORcnt.
  if (wave == 0) {
    unsigned long long gaddr =
        (unsigned long long)(Q + ((size_t)(b * V_ + v0)) * EH_);
    unsigned lds_off = (unsigned)(unsigned long long)(&sQ[0][0]);
    v4u g0 = {1u,                                     // count=1 (valid user D#)
              lds_off,                                // lds_addr
              (unsigned)(gaddr & 0xffffffffu),        // global_addr[31:0]
              (unsigned)((gaddr >> 32) & 0x01ffffffu) // global_addr[56:32]
                  | (2u << 30)};                      // type=2 ("image")
    v8i g1 = {(int)((2u << 16)    // data_size = 4 bytes
                    | (1u << 20)  // pad_enable
                    | (6u << 22)  // pad_interval: 128 dwords
                    | (3u << 25)),// pad_amount: 4 dwords
              (int)(128u << 16),  // tensor_dim0 = 128 (bits 79:48, low half)
              (int)(16u << 16),   // tensor_dim1 = 16  (bits 111:80, low half)
              (int)(128u << 16),  // tile_dim0 = 128   (bits 127:112)
              16,                 // tile_dim1 = 16    (bits 143:128)
              128,                // tensor_dim0_stride = 128 (bits 207:160)
              0, 0};
    v4i gz4 = {0, 0, 0, 0};
    v8i gz8 = {0, 0, 0, 0, 0, 0, 0, 0};
    __builtin_amdgcn_tensor_load_to_lds(g0, g1, gz4, gz4, gz8, 0);
    __builtin_amdgcn_s_wait_tensorcnt((unsigned short)0);
  }
#else
  for (int i = tid; i < 16 * (EH_ / 4); i += 256) {        // 512 float4s
    int r = i >> 5, c4 = (i & 31) << 2;
    *(float4*)(&sQ[r][c4]) =
        *(const float4*)(Q + ((size_t)(b * V_ + v0 + r)) * EH_ + c4);
  }
#endif
  for (int i = tid; i < EH_ / 4; i += 256)
    *(float4*)(&sRe[i << 2]) = *(const float4*)(eW1 + 128 * EH_ + (i << 2));
  for (int i = tid; i < 4 * (EH_ / 4); i += 256) {
    int o = i >> 5, c4 = (i & 31) << 2;
    *(float4*)(&sObs[o][c4]) = *(const float4*)(eW1 + (129 + o) * EH_ + c4);
  }
  __syncthreads();

  // preload eW2 B-fragments (K=128 split into 4 x 32), N padded 4 -> 16
  v16h bw[4];
#pragma unroll
  for (int w = 0; w < 4; ++w) {
#pragma unroll
    for (int i = 0; i < 16; ++i) {
      int k = 32 * w + hi * 8 + (i < 8 ? i : i + 8);
      bw[w][i] = (col < 4) ? w2h[k * 4 + col] : (_Float16)0.f;
    }
  }
  float bias = (col < 4) ? eb2[col] : 0.f;
  int m = col;  // edge row within v-tile

  for (int tu = 0; tu < 16; ++tu) {
    int u = uc * 128 + wave * 16 + tu;
    size_t erow = ((size_t)(b * V_ + u)) * V_ + v0 + m;
    float auv = A[erow];
    int tok = obs[erow];
    const float* obr = sObs[tok];
    const float* Pu = P + ((size_t)(b * V_ + u)) * EH_;  // includes T + eb1
    if (tu + 1 < 16) {  // prefetch next u (global_prefetch_b8)
      __builtin_prefetch(A + erow + V_, 0, 3);
      __builtin_prefetch(obs + erow + V_, 0, 3);
      __builtin_prefetch(Pu + EH_ + hi * 64, 0, 3);
    }
    v8f c = {bias, bias, bias, bias, bias, bias, bias, bias};
#pragma unroll
    for (int w = 0; w < 4; ++w) {
      int kb = 32 * w + hi * 8;
      float4 pA = *(const float4*)(Pu + kb);
      float4 pB = *(const float4*)(Pu + kb + 4);
      float4 pC = *(const float4*)(Pu + kb + 16);
      float4 pD = *(const float4*)(Pu + kb + 20);
      const float* qrow = sQ[m];
      float4 qA = *(const float4*)(qrow + kb);
      float4 qB = *(const float4*)(qrow + kb + 4);
      float4 qC = *(const float4*)(qrow + kb + 16);
      float4 qD = *(const float4*)(qrow + kb + 20);
      float4 rA = *(const float4*)(sRe + kb);
      float4 rB = *(const float4*)(sRe + kb + 4);
      float4 rC = *(const float4*)(sRe + kb + 16);
      float4 rD = *(const float4*)(sRe + kb + 20);
      float4 oA = *(const float4*)(obr + kb);
      float4 oB = *(const float4*)(obr + kb + 4);
      float4 oC = *(const float4*)(obr + kb + 16);
      float4 oD = *(const float4*)(obr + kb + 20);
      v16h af;
      ACT4(pA, qA, rA, oA, 0)
      ACT4(pB, qB, rB, oB, 4)
      ACT4(pC, qC, rC, oC, 8)
      ACT4(pD, qD, rD, oD, 12)
      c = wmma_f16_k32(af, bw[w], c);
    }
    if (col < 4) {
#pragma unroll
      for (int r = 0; r < 8; ++r) {
        out[(((size_t)(b * V_ + u)) * V_ + v0 + r + hi * 8) * 4 + col] = c[r];
      }
    }
  }
}

// ---------------------------------------------------------------------------
extern "C" void kernel_launch(void* const* d_in, const int* in_sizes, int n_in,
                              void* d_out, int out_size, void* d_ws, size_t ws_size,
                              hipStream_t stream) {
  const float* A   = (const float*)d_in[0];
  const float* nf  = (const float*)d_in[1];
  const int*   obs = (const int*)  d_in[2];
  const float* t   = (const float*)d_in[3];
  const float* tW1 = (const float*)d_in[4];
  const float* tb1 = (const float*)d_in[5];
  const float* tW2 = (const float*)d_in[6];
  const float* tb2 = (const float*)d_in[7];
  const float* npW = (const float*)d_in[8];
  const float* npb = (const float*)d_in[9];
  const float* gWs = (const float*)d_in[10];
  const float* gbs = (const float*)d_in[11];
  const float* gWn = (const float*)d_in[12];
  const float* gbn = (const float*)d_in[13];
  const float* gG  = (const float*)d_in[14];
  const float* gB  = (const float*)d_in[15];
  const float* eW1 = (const float*)d_in[16];
  const float* eb1 = (const float*)d_in[17];
  const float* eW2 = (const float*)d_in[18];
  const float* eb2 = (const float*)d_in[19];

  float* ws   = (float*)d_ws;
  float* temb = ws + 0;        // B*64
  float* tvec = ws + 128;      // B*128
  float* dis  = ws + 384;      // B*512
  float* An   = ws + 1408;     // B*V*V
  float* h0   = ws + 525696;   // B*V*64
  float* h1   = ws + 591232;   // B*V*64
  float* P    = ws + 656768;   // B*V*128
  float* Q    = ws + 787840;   // B*V*128
  _Float16* w2h = (_Float16*)(ws + 918912);  // 512 halves
  float* out = (float*)d_out;

  k_setup<<<1, 256, 0, stream>>>(t, tW1, tb1, tW2, tb2, eW1, eb1, eW2, temb, tvec, w2h);
  k_hinit<<<(B_ * V_ * HID_) / 256, 256, 0, stream>>>(nf, npW, npb, temb, h0);
  k_dis<<<B_ * V_, 128, 0, stream>>>(A, dis);
  k_anorm<<<(B_ * V_ * V_) / 256, 256, 0, stream>>>(A, dis, An);
  k_gnn<<<B_ * (V_ / 16), 128, 0, stream>>>(h0, h1, An, gWs + 0 * 4096, gbs + 0,
                                            gWn + 0 * 4096, gbn + 0, gG + 0, gB + 0);
  k_gnn<<<B_ * (V_ / 16), 128, 0, stream>>>(h1, h0, An, gWs + 1 * 4096, gbs + 64,
                                            gWn + 1 * 4096, gbn + 64, gG + 64, gB + 64);
  k_gnn<<<B_ * (V_ / 16), 128, 0, stream>>>(h0, h1, An, gWs + 2 * 4096, gbs + 128,
                                            gWn + 2 * 4096, gbn + 128, gG + 128, gB + 128);
  k_pq<<<B_ * (V_ / 16), 256, 0, stream>>>(h1, eW1, tvec, P, Q);
  k_edge<<<256, 256, 0, stream>>>(A, obs, P, Q, eW1, w2h, eb2, out);
}
